// DynamicsLocallyLinear_20512763806123
// MI455X (gfx1250) — matrix operand order repaired and checked by reading, not compiled, not run on hardware
//
#include <hip/hip_runtime.h>
#include <hip/hip_bf16.h>

typedef __attribute__((ext_vector_type(16))) _Float16 v16h;
typedef __attribute__((ext_vector_type(8)))  _Float16 v8h;
typedef __attribute__((ext_vector_type(8)))  float    v8f;

#define HD     256
#define LT     16
#define NTOK   65536            // B*T = 32*2048
#define NTILES 4096             // NTOK / 16 tokens per block

// f16 weight workspace layout (element offsets)
#define OFF_WZH 0
#define N_WZH   (256*16)
#define OFF_WHH (N_WZH)
#define N_WHH   (768*256)
#define OFF_WHZ (OFF_WHH + N_WHH)
#define N_WHZ   (272*256)
#define WS_HALFS (OFF_WHZ + N_WHZ)

#define WMMA_F16(A,B,C) __builtin_amdgcn_wmma_f32_16x16x32_f16(false,(A),false,(B),(short)0,(C),false,false)

__device__ __forceinline__ float sigm(float x) { return 1.0f / (1.0f + __expf(-x)); }

// ---------------------------------------------------------------------------
// Weight precision prep: fp32 -> f16, row-major [N][K] preserved so that
// B-fragments (lane: N = n0+lane%16, K = kb*32 + (lane/16)*16 + i) are one
// contiguous 32-byte global load per lane.
// ---------------------------------------------------------------------------
__global__ void prep_weights_f16(const float* __restrict__ Wzh,
                                 const float* __restrict__ Whh,
                                 const float* __restrict__ Whz,
                                 _Float16* __restrict__ ws)
{
  int i = blockIdx.x * blockDim.x + threadIdx.x;
  if (i >= WS_HALFS) return;
  float v;
  if (i < OFF_WHH)      v = Wzh[i];
  else if (i < OFF_WHZ) v = Whh[i - OFF_WHH];
  else                  v = Whz[i - OFF_WHZ];
  ws[i] = (_Float16)v;
}

// A-fragment (contiguous k order, same as the B-fragment loads) from an LDS-staged f16 row.
// rowbase = &act[M*256]; halfs i -> K = kb*32 + lh*16 + i.
__device__ __forceinline__ v16h load_a_frag(const _Float16* rowbase, int kb, int lh)
{
  const v8h a0 = *(const v8h*)(rowbase + kb*32 + lh*16);
  const v8h a1 = *(const v8h*)(rowbase + kb*32 + lh*16 + 8);
  v16h a;
#pragma unroll
  for (int i = 0; i < 8; i++) { a[i] = a0[i]; a[8+i] = a1[i]; }
  return a;
}

// ---------------------------------------------------------------------------
// Fused kernel: one block = 16 tokens.
//   Phase 1: h = z @ Wzh^T + b_zh          (WMMA, K=16 zero-padded to 32)
//   Phase 2: gh = h @ Whh^T + b_hh         (WMMA, K=256, accs in registers)
//   Phase 3: GRU gates in-register -> h_out (f16 to LDS)
//   Phase 4: A_flat = h_out @ Whz^T + b_hz (WMMA, 17 col-blocks)
//   Phase 5: Az = z @ W_A^T in exact fp32 VALU (output 1)
//   Phase 6: per-token 16x16 one-sided Jacobi SVD (2 tokens/wave, XOR pairing),
//            z_out = U diag(sigmoid(s)) V^T z with sorted-sigma rank matching
// ---------------------------------------------------------------------------
__launch_bounds__(256)
__global__ void dyn_fused(const float* __restrict__ z,
                          const float* __restrict__ b_zh,
                          const float* __restrict__ b_ih,
                          const float* __restrict__ b_hh,
                          const float* __restrict__ b_hz,
                          const float* __restrict__ W_A,
                          const _Float16* __restrict__ ws16,
                          float* __restrict__ out_Az,
                          float* __restrict__ out_zout)
{
  __shared__ __align__(16) unsigned char smem[35328];
  float*    zf32   = (float*)(smem);                 // [16][16] f32, persists
  _Float16* zf16   = (_Float16*)(smem + 1024);       // [16][16] f16
  _Float16* h16    = (_Float16*)(smem + 1536);       // [16][256] f16 (phases 1-2)
  _Float16* hout16 = (_Float16*)(smem + 1536 + 8192);// [16][256] f16 (phases 3-4)
  float*    svdbuf = (float*)(smem + 1536);          // [16][16][16] f32, aliases h16+hout16 (phase 6)
  float*    h32    = (float*)(smem + 17920);         // [16][256] f32 (phases 1-3)
  float*    aflat  = (float*)(smem + 17920);         // [16][272] f32, aliases h32 (phases 4-6)

  const int tid  = threadIdx.x;
  const int wave = tid >> 5;
  const int lane = tid & 31;
  const int lm   = lane & 15;   // N (or M) index within fragment
  const int lh   = lane >> 4;   // lane-half selector
  const int tok0 = blockIdx.x * 16;

  // ---- Phase 0: stage z tile ----
  {
    float v = z[tok0 * LT + tid];
    zf32[tid] = v;
    zf16[tid] = (_Float16)v;
  }
  __syncthreads();

  // ---- Phase 1: h = z @ Wzh^T + b_zh (K=16 padded to 32) ----
  v16h aZ;
  {
    const v8h lo = *(const v8h*)&zf16[lm * LT + lh * 8];   // real K = lh*8 + i
#pragma unroll
    for (int i = 0; i < 8; i++) { aZ[i] = lo[i]; aZ[8+i] = (_Float16)0.0f; } // K>=16 pad
  }
#pragma unroll
  for (int j = 0; j < 2; j++) {
    int nidx = (wave * 2 + j) * 16 + lm;
    v16h b;
#pragma unroll
    for (int i = 0; i < 16; i++) b[i] = (_Float16)0.0f;     // elements 8-15: K=16..31 pad
    {
      const v8h b0 = *(const v8h*)&ws16[OFF_WZH + nidx * LT + lh * 8];
#pragma unroll
      for (int i = 0; i < 8; i++) { b[i] = b0[i]; }
    }
    float bias = b_zh[nidx];
    v8f c;
#pragma unroll
    for (int v = 0; v < 8; v++) c[v] = bias;
    c = WMMA_F16(aZ, b, c);
#pragma unroll
    for (int v = 0; v < 8; v++) {
      int m = v + 8 * lh;                                   // C layout: M = v + 8*(lane/16)
      h32[m * HD + nidx] = c[v];
      h16[m * HD + nidx] = (_Float16)c[v];
    }
  }
  __syncthreads();

  // ---- Phase 2: gh = h @ Whh^T + b_hh ; 6 accumulators per wave ----
  v8f acc[6];
#pragma unroll
  for (int j = 0; j < 6; j++) {
    float bias = b_hh[(wave + 8 * j) * 16 + lm];
#pragma unroll
    for (int v = 0; v < 8; v++) acc[j][v] = bias;
  }
  for (int kb = 0; kb < 8; kb++) {
    v16h a = load_a_frag(&h16[lm * HD], kb, lh);
#pragma unroll
    for (int j = 0; j < 6; j++) {
      int nidx = (wave + 8 * j) * 16 + lm;
      const _Float16* bp = &ws16[OFF_WHH + nidx * HD + kb * 32 + lh * 16];
      if (kb < 7) __builtin_prefetch(bp + 32, 0, 1);        // global_prefetch_b8
      v16h b = *(const v16h*)bp;
      acc[j] = WMMA_F16(a, b, acc[j]);
    }
  }

  // ---- Phase 3: GRU gates in-register (acc[jj]=hr, acc[jj+2]=hu, acc[jj+4]=hn) ----
#pragma unroll
  for (int jj = 0; jj < 2; jj++) {
    int nidx = (wave + 8 * jj) * 16 + lm;
    float ir  = b_ih[nidx];
    float iu  = b_ih[HD + nidx];
    float inn = b_ih[2 * HD + nidx];
#pragma unroll
    for (int v = 0; v < 8; v++) {
      int m = v + 8 * lh;
      float hv = h32[m * HD + nidx];
      float r  = sigm(ir + acc[jj][v]);
      float u  = sigm(iu + acc[jj + 2][v]);
      float x  = inn + r * acc[jj + 4][v];
      float nn = 2.0f * sigm(2.0f * x) - 1.0f;              // tanh
      float ho = (1.0f - u) * nn + u * hv;
      hout16[m * HD + nidx] = (_Float16)ho;
    }
  }
  __syncthreads();  // h32 dead beyond here; aflat may overwrite region

  // ---- Phase 4: A_flat = h_out @ Whz^T + b_hz (17 col blocks) ----
  for (int cb = wave; cb < 17; cb += 8) {
    int nidx = cb * 16 + lm;
    float bias = b_hz[nidx];
    v8f c;
#pragma unroll
    for (int v = 0; v < 8; v++) c[v] = bias;
    for (int kb = 0; kb < 8; kb++) {
      v16h a = load_a_frag(&hout16[lm * HD], kb, lh);
      v16h b = *(const v16h*)&ws16[OFF_WHZ + nidx * HD + kb * 32 + lh * 16];
      c = WMMA_F16(a, b, c);
    }
#pragma unroll
    for (int v = 0; v < 8; v++) {
      int m = v + 8 * lh;
      aflat[m * 272 + nidx] = c[v];
    }
  }

  // ---- Phase 5: Az = z @ W_A^T, exact fp32 (output half 1) ----
  {
    int m = tid >> 4, jc = tid & 15;
    float s = 0.0f;
#pragma unroll
    for (int k = 0; k < 16; k++) s += zf32[m * 16 + k] * W_A[jc * 16 + k];
    out_Az[(tok0 + m) * LT + jc] = s;
  }
  __syncthreads();  // aflat complete; h16/hout16 dead -> svdbuf usable

  // ---- Phase 6: batched 16x16 one-sided Jacobi SVD, 2 tokens per wave ----
  const int tokl = 2 * wave + lh;       // token in tile (lanes 0-15 / 16-31 split)
  const int ccol = lm;                  // my column of W = A*V and of V
  float Wc[16], Vc[16], zi[16];
#pragma unroll
  for (int i = 0; i < 16; i++) {
    Wc[i] = aflat[tokl * 272 + 16 + i * 16 + ccol];   // A[i][c]
    Vc[i] = (i == ccol) ? 1.0f : 0.0f;
    zi[i] = zf32[tokl * 16 + i];
  }
  float sval = sigm(aflat[tokl * 272 + ccol]);        // sigmoid gate values

#pragma unroll 1
  for (int sweep = 0; sweep < 8; sweep++) {
#pragma unroll 1
    for (int r = 1; r < 16; r++) {                    // XOR pairing: all 120 pairs/sweep
      float Wp[16], Vp[16];
#pragma unroll
      for (int i = 0; i < 16; i++) {
        Wp[i] = __shfl_xor(Wc[i], r, 32);
        Vp[i] = __shfl_xor(Vc[i], r, 32);
      }
      float nself = 0.0f, gam = 0.0f;
#pragma unroll
      for (int i = 0; i < 16; i++) {
        nself += Wc[i] * Wc[i];
        gam   += Wc[i] * Wp[i];                       // symmetric: identical on both lanes
      }
      float nother = __shfl_xor(nself, r, 32);
      bool  isP    = ccol < (ccol ^ r);
      float alpha  = isP ? nself : nother;
      float beta   = isP ? nother : nself;
      float cr = 1.0f, sr = 0.0f;
      if (fabsf(gam) > 1e-20f) {
        float zeta = (beta - alpha) / (2.0f * gam);
        float t    = copysignf(1.0f, zeta) / (fabsf(zeta) + sqrtf(1.0f + zeta * zeta));
        cr = 1.0f / sqrtf(1.0f + t * t);
        sr = t * cr;
      }
#pragma unroll
      for (int i = 0; i < 16; i++) {
        float wn = isP ? (cr * Wc[i] - sr * Wp[i]) : (sr * Wp[i] + cr * Wc[i]);
        float vn = isP ? (cr * Vc[i] - sr * Vp[i]) : (sr * Vp[i] + cr * Vc[i]);
        Wc[i] = wn; Vc[i] = vn;
      }
    }
  }

  // sigma, rank (descending-sort index to match jnp.linalg.svd ordering)
  float sig2 = 0.0f;
#pragma unroll
  for (int i = 0; i < 16; i++) sig2 += Wc[i] * Wc[i];
  float sigma = sqrtf(sig2);
  int base = lane & 16;
  int rank = 0;
#pragma unroll
  for (int d = 0; d < 16; d++) {
    float sd = __shfl(sigma, base + d, 32);
    rank += (sd > sigma) || ((sd == sigma) && (d < ccol));
  }
  float smine = __shfl(sval, base + rank, 32);        // s for my sorted slot
  float tdot = 0.0f;
#pragma unroll
  for (int i = 0; i < 16; i++) tdot += Vc[i] * zi[i]; // (V^T z)_c
  float inv  = (sigma > 1e-20f) ? (1.0f / sigma) : 0.0f;
  float coef = smine * tdot * inv;                    // z_out = sum_c coef_c * W_c

#pragma unroll
  for (int i = 0; i < 16; i++)
    svdbuf[tokl * 256 + i * 16 + ccol] = coef * Wc[i];
  __syncthreads();
  {
    const float4* row = (const float4*)&svdbuf[tokl * 256 + lm * 16];
    float4 r0 = row[0], r1 = row[1], r2 = row[2], r3 = row[3];
    float sum = (r0.x + r0.y + r0.z + r0.w) + (r1.x + r1.y + r1.z + r1.w)
              + (r2.x + r2.y + r2.z + r2.w) + (r3.x + r3.y + r3.z + r3.w);
    out_zout[(tok0 + tokl) * LT + lm] = sum;
  }
}

extern "C" void kernel_launch(void* const* d_in, const int* in_sizes, int n_in,
                              void* d_out, int out_size, void* d_ws, size_t ws_size,
                              hipStream_t stream)
{
  (void)in_sizes; (void)n_in; (void)out_size; (void)ws_size;
  const float* z   = (const float*)d_in[0];
  const float* Wzh = (const float*)d_in[1];
  const float* bzh = (const float*)d_in[2];
  const float* Whh = (const float*)d_in[3];
  const float* bih = (const float*)d_in[4];
  const float* bhh = (const float*)d_in[5];
  const float* Whz = (const float*)d_in[6];
  const float* bhz = (const float*)d_in[7];
  const float* WA  = (const float*)d_in[8];
  float* out = (float*)d_out;
  _Float16* ws16 = (_Float16*)d_ws;   // needs 540,672 bytes

  prep_weights_f16<<<(WS_HALFS + 255) / 256, 256, 0, stream>>>(Wzh, Whh, Whz, ws16);
  dyn_fused<<<NTILES, 256, 0, stream>>>(z, bzh, bih, bhh, bhz, WA, ws16,
                                        out, out + (size_t)NTOK * LT);
}
